// StandardAttention_51634096833047
// MI455X (gfx1250) — matrix operand order repaired and checked by reading, not compiled, not run on hardware
//
#include <hip/hip_runtime.h>
#include <hip/hip_bf16.h>

typedef __attribute__((ext_vector_type(16))) __bf16 v16bf;
typedef __attribute__((ext_vector_type(8)))  float  v8f;

#define DIM      1024
#define HEADS    16
#define HEAD_DIM 64
#define BATCH    2
#define SEQ      2048
#define MROWS    (BATCH * SEQ)   // 4096
#define PROW     36              // padded LDS row stride (floats): conflict-free, 16B aligned

// A-matrix (16x32 bf16) per-lane K mapping used here (contiguous order, matched by B loads):
// lane L holds row M=L&15; element i -> k = a_kbase(i) + (L>>4)*16
__device__ __forceinline__ int a_kbase(int i) {
    return i;   // {0..15}
}
__device__ __forceinline__ int a_kmap(int i, int hi) { return a_kbase(i) + hi * 16; }

__device__ __forceinline__ v8f wmma_bf16(v16bf a, v16bf b, v8f c) {
    return __builtin_amdgcn_wmma_f32_16x16x32_bf16(false, a, false, b, (short)0, c,
                                                   false, false);
}

__global__ void cast_bf16_kernel(const float* __restrict__ in,
                                 __bf16* __restrict__ out, int n) {
    int i = blockIdx.x * blockDim.x + threadIdx.x;
    if (i < n) out[i] = (__bf16)in[i];
}

// W [K=1024][N=1024] row-major f32 -> WT [N][K] bf16 (contiguous columns),
// optionally pre-scaled (folds 1/sqrt(HEAD_DIM) into Wq for free).
__global__ void transpose_cast_kernel(const float* __restrict__ in,
                                      __bf16* __restrict__ out, float scale) {
    int idx = blockIdx.x * blockDim.x + threadIdx.x;  // 1M elements
    int k = idx >> 10, n = idx & 1023;
    out[(size_t)n * DIM + k] = (__bf16)(in[idx] * scale);
}

// C[M,N] = A[M,K](bf16 row-major) x WT[N,K](bf16). Wave tile: 32 rows x 64 cols.
// MODE 0: f32 out row-major; MODE 1: bf16 out row-major;
// MODE 2: bf16 out head-transposed Vt[b][h][dh][s] (s contiguous)
template <int MODE>
__global__ __launch_bounds__(128) void gemm_bf16_kernel(
        const __bf16* __restrict__ A, const __bf16* __restrict__ WT,
        void* __restrict__ Cout, int M, int N, int K) {
    const int lane = threadIdx.x & 31;
    const int wave = threadIdx.x >> 5;
    const int hi = lane >> 4;
    const int ln = lane & 15;
    const int m0 = __builtin_amdgcn_readfirstlane((blockIdx.y * 4 + wave) * 32);
    const int n0 = blockIdx.x * 64;
    if (m0 >= M) return;

    v8f acc[2][4] = {};
    for (int k0 = 0; k0 < K; k0 += 32) {
        v16bf a[2];
        #pragma unroll
        for (int mi = 0; mi < 2; ++mi)
            #pragma unroll
            for (int i = 0; i < 16; ++i)
                a[mi][i] = A[(size_t)(m0 + mi * 16 + ln) * K + k0 + a_kmap(i, hi)];
        #pragma unroll
        for (int t = 0; t < 4; ++t) {
            v16bf b;
            const __bf16* wp = WT + (size_t)(n0 + t * 16 + ln) * K + k0 + hi * 16;
            #pragma unroll
            for (int i = 0; i < 16; ++i) b[i] = wp[i];
            #pragma unroll
            for (int mi = 0; mi < 2; ++mi)
                acc[mi][t] = wmma_bf16(a[mi], b, acc[mi][t]);
        }
    }
    #pragma unroll
    for (int mi = 0; mi < 2; ++mi) {
        #pragma unroll
        for (int t = 0; t < 4; ++t) {
            #pragma unroll
            for (int r = 0; r < 8; ++r) {
                const int m = m0 + mi * 16 + r + hi * 8;
                const int n = n0 + t * 16 + ln;
                const float v = acc[mi][t][r];
                if (MODE == 0) {
                    ((float*)Cout)[(size_t)m * N + n] = v;
                } else if (MODE == 1) {
                    ((__bf16*)Cout)[(size_t)m * N + n] = (__bf16)v;
                } else {
                    // m = b*SEQ + s ; n = h*HEAD_DIM + dh
                    const int bb = m >> 11, ss = m & (SEQ - 1);
                    const int hh = n >> 6, dd = n & (HEAD_DIM - 1);
                    ((__bf16*)Cout)[(((size_t)bb * HEADS + hh) * HEAD_DIM + dd) * SEQ + ss] =
                        (__bf16)v;
                }
            }
        }
    }
}

// Flash attention: one wave per (b, h, 16-row q tile). Keys in 32-wide blocks.
// Q (pre-scaled by 1/sqrt(Dh)), K in [B,S,H*Dh] bf16; V head-transposed
// Vt[b][h][dh][s] bf16. Row-owner softmax: lane (L&15) owns score row L&15.
__global__ __launch_bounds__(128) void flash_attn_kernel(
        const __bf16* __restrict__ Q, const __bf16* __restrict__ Kmat,
        const __bf16* __restrict__ Vt, __bf16* __restrict__ O) {
    __shared__ float plds[4][16 * PROW];  // per-wave raw-S staging
    const int lane = threadIdx.x & 31;
    const int wave = threadIdx.x >> 5;
    const int hi = lane >> 4;
    const int ln = lane & 15;
    const int b = blockIdx.x >> 4;
    const int h = blockIdx.x & 15;
    // wave-uniform -> force into SGPRs so the kb loop control stays scalar
    const int q0 = __builtin_amdgcn_readfirstlane((blockIdx.y * 4 + wave) * 16);
    const size_t base  = (size_t)b * SEQ * DIM + (size_t)h * HEAD_DIM;  // Q/K/O
    const size_t vbase = ((size_t)b * HEADS + h) * HEAD_DIM * SEQ;      // Vt

    // Q A-fragments, two 32-wide dh halves (scale already folded into Wq)
    v16bf qa[2];
    #pragma unroll
    for (int d = 0; d < 2; ++d)
        #pragma unroll
        for (int i = 0; i < 16; ++i)
            qa[d][i] = Q[base + (size_t)(q0 + ln) * DIM + d * 32 + a_kmap(i, hi)];

    v8f o[4] = {};
    float msc = -1e30f;  // running max of owned row (q0 + ln)
    float lsc = 0.0f;    // running sum of owned row

    const int nkb = (q0 + 16 + 31) >> 5;  // causal bound, 32-key blocks (scalar)
    for (int kb = 0; kb < nkb; ++kb) {
        const int key0 = kb * 32;
        if (kb + 1 < nkb) {  // scalar branch: overlap next K block fetch
            __builtin_prefetch(Kmat + base + (size_t)(key0 + 32 + lane) * DIM, 0, 3);
        }
        // S tiles: keys [key0,key0+16) and [key0+16,key0+32)
        v8f s0 = {}, s1 = {};
        #pragma unroll
        for (int d = 0; d < 2; ++d) {
            v16bf bk0, bk1;
            const __bf16* kp0 = Kmat + base + (size_t)(key0 + ln) * DIM + d * 32 + hi * 16;
            const __bf16* kp1 = kp0 + (size_t)16 * DIM;
            #pragma unroll
            for (int i = 0; i < 16; ++i) { bk0[i] = kp0[i]; bk1[i] = kp1[i]; }
            s0 = wmma_bf16(qa[d], bk0, s0);
            s1 = wmma_bf16(qa[d], bk1, s1);
        }
        // Stage raw S (C-layout: row = r + hi*8, col = ln / ln+16)
        #pragma unroll
        for (int r = 0; r < 8; ++r) {
            plds[wave][(r + hi * 8) * PROW + ln]      = s0[r];
            plds[wave][(r + hi * 8) * PROW + 16 + ln] = s1[r];
        }
        asm volatile("s_wait_dscnt 0" ::: "memory");  // LDS in-order per wave
        // Owner lane processes its full row in registers
        const float* rp = &plds[wave][ln * PROW];
        float sv[32];
        #pragma unroll
        for (int j = 0; j < 32; ++j) sv[j] = rp[j];
        if (q0 < key0 + 31) {  // diagonal block (scalar branch): causal mask
            const int thr = q0 + ln - key0;  // cols j <= thr are valid (thr >= 0 always)
            #pragma unroll
            for (int j = 1; j < 32; ++j)
                sv[j] = (j <= thr) ? sv[j] : -1e30f;
        }
        float rmax = sv[0];
        #pragma unroll
        for (int j = 1; j < 32; ++j) rmax = fmaxf(rmax, sv[j]);
        const float mnew = fmaxf(msc, rmax);
        const float alpha = __expf(msc - mnew);
        msc = mnew;
        float psum = 0.0f;
        #pragma unroll
        for (int j = 0; j < 32; ++j) { sv[j] = __expf(sv[j] - mnew); psum += sv[j]; }
        lsc = lsc * alpha + psum;
        // P A-fragment from registers; static indices + one cndmask per element
        v16bf pa;
        #pragma unroll
        for (int i = 0; i < 16; ++i) {
            const int kb0 = a_kbase(i);               // compile-time
            const float pv = hi ? sv[kb0 + 16] : sv[kb0];
            pa[i] = (__bf16)pv;
        }
        // Broadcast alpha from owner lanes to C-layout rows (8 bpermutes)
        #pragma unroll
        for (int r = 0; r < 8; ++r) {
            const float ar = __shfl(alpha, (hi << 4) + r + (hi << 3), 32);
            #pragma unroll
            for (int t = 0; t < 4; ++t) o[t][r] *= ar;
        }
        // O += P x V : Vt gives contiguous 16-key runs per dh column
        #pragma unroll
        for (int t = 0; t < 4; ++t) {
            v16bf vb;
            const __bf16* vp = Vt + vbase + (size_t)(t * 16 + ln) * SEQ + key0 + hi * 16;
            #pragma unroll
            for (int i = 0; i < 16; ++i) vb[i] = vp[i];
            o[t] = wmma_bf16(pa, vb, o[t]);
        }
    }
    // Normalize (broadcast 1/l from owner lanes) and store bf16 into [B,S,H*Dh]
    #pragma unroll
    for (int r = 0; r < 8; ++r) {
        const float lr = __shfl(lsc, (hi << 4) + r + (hi << 3), 32);
        const float inv = 1.0f / lr;
        const size_t rowoff = base + (size_t)(q0 + r + hi * 8) * DIM;
        #pragma unroll
        for (int t = 0; t < 4; ++t)
            O[rowoff + t * 16 + ln] = (__bf16)(o[t][r] * inv);
    }
}

extern "C" void kernel_launch(void* const* d_in, const int* in_sizes, int n_in,
                              void* d_out, int out_size, void* d_ws, size_t ws_size,
                              hipStream_t stream) {
    const float* x  = (const float*)d_in[0];
    // d_in[1] = mask: unused, causality computed analytically
    const float* Wq = (const float*)d_in[2];
    const float* Wk = (const float*)d_in[3];
    const float* Wv = (const float*)d_in[4];
    const float* Wo = (const float*)d_in[5];
    float* out = (float*)d_out;

    char* ws = (char*)d_ws;
    const size_t SZ_ACT = (size_t)MROWS * DIM * sizeof(__bf16);  // 8 MB
    const size_t SZ_W   = (size_t)DIM * DIM * sizeof(__bf16);    // 2 MB
    __bf16* xb  = (__bf16*)(ws);
    __bf16* Qb  = (__bf16*)(ws + 1 * SZ_ACT);
    __bf16* Kb  = (__bf16*)(ws + 2 * SZ_ACT);
    __bf16* Vtb = (__bf16*)(ws + 3 * SZ_ACT);
    __bf16* AOb = (__bf16*)(ws + 4 * SZ_ACT);
    __bf16* WqT = (__bf16*)(ws + 5 * SZ_ACT);
    __bf16* WkT = (__bf16*)(ws + 5 * SZ_ACT + 1 * SZ_W);
    __bf16* WvT = (__bf16*)(ws + 5 * SZ_ACT + 2 * SZ_W);
    __bf16* WoT = (__bf16*)(ws + 5 * SZ_ACT + 3 * SZ_W);

    const int nact = MROWS * DIM;
    cast_bf16_kernel<<<(nact + 255) / 256, 256, 0, stream>>>(x, xb, nact);
    const int nw = DIM * DIM;
    const float qscale = 0.125f;  // 1/sqrt(HEAD_DIM), folded into Wq
    transpose_cast_kernel<<<nw / 256, 256, 0, stream>>>(Wq, WqT, qscale);
    transpose_cast_kernel<<<nw / 256, 256, 0, stream>>>(Wk, WkT, 1.0f);
    transpose_cast_kernel<<<nw / 256, 256, 0, stream>>>(Wv, WvT, 1.0f);
    transpose_cast_kernel<<<nw / 256, 256, 0, stream>>>(Wo, WoT, 1.0f);

    dim3 gg(DIM / 64, MROWS / 128);  // 16 x 32 blocks, 4 waves x (32x64 tile)
    gemm_bf16_kernel<1><<<gg, 128, 0, stream>>>(xb, WqT, Qb,  MROWS, DIM, DIM);
    gemm_bf16_kernel<1><<<gg, 128, 0, stream>>>(xb, WkT, Kb,  MROWS, DIM, DIM);
    gemm_bf16_kernel<2><<<gg, 128, 0, stream>>>(xb, WvT, Vtb, MROWS, DIM, DIM);

    dim3 ga(BATCH * HEADS, SEQ / 16 / 4);  // 32 x 32 blocks, 4 q-tiles each
    flash_attn_kernel<<<ga, 128, 0, stream>>>(Qb, Kb, Vtb, AOb);

    gemm_bf16_kernel<0><<<gg, 128, 0, stream>>>(AOb, WoT, out, MROWS, DIM, DIM);
}
